// MFFNet_61495341744155
// MI455X (gfx1250) — matrix operand order repaired and checked by reading, not compiled, not run on hardware
//
#include <hip/hip_runtime.h>
#include <math.h>

// ---------------------------------------------------------------- types ----
typedef _Float16 f16;
typedef __attribute__((ext_vector_type(16))) _Float16 v16h;
typedef __attribute__((ext_vector_type(8)))  float    v8f;

struct alignas(16) U4 { unsigned int x, y, z, w; };
union FragH { U4 u[2]; v16h h; };
union H2   { f16 h[2]; unsigned int u; };
union H8   { f16 h[8]; U4 u; };

#define N_BATCH 64
#define T_LEN   300
#define V_N     18
#define M_P     2
#define B_TOT   128                         // N*M
#define BT_TOT  (B_TOT * T_LEN)             // 38400
#define MROWS   ((long)B_TOT * T_LEN * V_N) // 691200 rows, multiple of 256
#define EPSF    1e-5f

// ------------------------------------------------------- weight packing ----
// Pack W (N x C, row-major, f32) into WMMA-B fragment order (f16):
// frag = kp*(N/16)+np ; per lane 8 dwords, dword v = half2{ B[k0][n], B[k0+1][n] }
// with n = np*16 + (lane&15), k0 = kp*32 + ((lane&16)?16:0) + 2v, B[k][n] = W[n][k].
__global__ void pack_wmma_b(const float* __restrict__ W, int ldw, int col0,
                            int cvalid, int kpad, int n,
                            unsigned int* __restrict__ out) {
    int tid = blockIdx.x * blockDim.x + threadIdx.x;
    int total = (kpad >> 5) * (n >> 4) * 32;
    if (tid >= total) return;
    int lane = tid & 31;
    int frag = tid >> 5;
    int col = (frag % (n >> 4)) * 16 + (lane & 15);
    int kp  = frag / (n >> 4);
    int kbase = kp * 32 + ((lane & 16) ? 16 : 0);
    unsigned int* o = out + ((size_t)frag * 32 + lane) * 8;
#pragma unroll
    for (int v = 0; v < 8; ++v) {
        int k0 = kbase + 2 * v;
        float f0 = (k0     < cvalid) ? W[(size_t)col * ldw + col0 + k0    ] : 0.0f;
        float f1 = (k0 + 1 < cvalid) ? W[(size_t)col * ldw + col0 + k0 + 1] : 0.0f;
        H2 p; p.h[0] = (f16)f0; p.h[1] = (f16)f1;
        o[v] = p.u;
    }
}

// ------------------------------------------------------------ WMMA GEMM ----
// C[M,N] (f32) = A[M,kpad] (f16, row stride lda) * Bpacked + bias  (or += if accum)
// Block: 256 thr = 8 waves, wave tile 32x64 (2 M-subtiles x 4 N-subtiles).
__global__ __launch_bounds__(256) void wmma_gemm_f16(
    const f16* __restrict__ A, int lda,
    const U4*  __restrict__ Bp,
    const float* __restrict__ bias,
    float* __restrict__ C, int ldc,
    int kpad, int n, int accum)
{
    int lane = threadIdx.x & 31;
    int wave = threadIdx.x >> 5;
    long rowBase = (long)blockIdx.x * 256 + wave * 32;
    int  colBase = blockIdx.y * 64;
    int  l15 = lane & 15;
    int  hi  = (lane & 16) ? 1 : 0;

    v8f acc[2][4];
    if (accum) {
#pragma unroll
        for (int i = 0; i < 2; ++i)
#pragma unroll
            for (int j = 0; j < 4; ++j)
#pragma unroll
                for (int r = 0; r < 8; ++r)
                    acc[i][j][r] = C[(rowBase + i * 16 + hi * 8 + r) * ldc +
                                     colBase + j * 16 + l15];
    } else {
#pragma unroll
        for (int j = 0; j < 4; ++j) {
            float bv = bias ? bias[colBase + j * 16 + l15] : 0.0f;
#pragma unroll
            for (int i = 0; i < 2; ++i)
#pragma unroll
                for (int r = 0; r < 8; ++r) acc[i][j][r] = bv;
        }
    }

    int kIters = kpad >> 5;
    int nfr    = n >> 4;
    int aoff   = hi * 32;
    for (int kp = 0; kp < kIters; ++kp) {
        FragH b[4];
#pragma unroll
        for (int j = 0; j < 4; ++j) {
            const U4* bp = Bp + ((size_t)(kp * nfr + (colBase >> 4) + j) * 32 + lane) * 2;
            b[j].u[0] = bp[0];
            b[j].u[1] = bp[1];
        }
        FragH a[2];
#pragma unroll
        for (int i = 0; i < 2; ++i) {
            const char* p = (const char*)(A + (rowBase + i * 16 + l15) * lda) + kp * 64;
            a[i].u[0] = *(const U4*)(p + aoff);
            a[i].u[1] = *(const U4*)(p + 16 + aoff);
            __builtin_prefetch(p + 64, 0, 1);   // next K-panel -> global_prefetch
        }
#pragma unroll
        for (int i = 0; i < 2; ++i)
#pragma unroll
            for (int j = 0; j < 4; ++j)
                acc[i][j] = __builtin_amdgcn_wmma_f32_16x16x32_f16(
                    false, a[i].h, false, b[j].h, (short)0, acc[i][j], false, false);
    }

#pragma unroll
    for (int i = 0; i < 2; ++i)
#pragma unroll
        for (int j = 0; j < 4; ++j)
#pragma unroll
            for (int r = 0; r < 8; ++r)
                C[(rowBase + i * 16 + hi * 8 + r) * ldc + colBase + j * 16 + l15] =
                    acc[i][j][r];
}

// --------------------------------------------------------------- helpers ----
__global__ void zero_f32(float* p, int n) {
    int i = blockIdx.x * blockDim.x + threadIdx.x;
    if (i < n) p[i] = 0.0f;
}

// data_bn statistics: one block per feature f = v*3+c, reduce over N*T*M.
__global__ void dbn_stats(const float* __restrict__ pos, float* __restrict__ mv,
                          float* __restrict__ vv) {
    __shared__ float rs[256], rq[256];
    int f = blockIdx.x, v = f / 3, c = f % 3;
    float s = 0.f, q = 0.f;
    for (int j = threadIdx.x; j < N_BATCH * T_LEN * M_P; j += 256) {
        int n = j / (T_LEN * M_P);
        int r = j - n * (T_LEN * M_P);
        int t = r >> 1, m = r & 1;
        float x = pos[(long)n * 32400 + (long)c * 10800 + t * 36 + v * 2 + m];
        s += x; q += x * x;
    }
    rs[threadIdx.x] = s; rq[threadIdx.x] = q;
    __syncthreads();
    for (int st = 128; st > 0; st >>= 1) {
        if (threadIdx.x < st) {
            rs[threadIdx.x] += rs[threadIdx.x + st];
            rq[threadIdx.x] += rq[threadIdx.x + st];
        }
        __syncthreads();
    }
    if (threadIdx.x == 0) {
        float cnt = (float)(N_BATCH * T_LEN * M_P);
        float mean = rs[0] / cnt;
        mv[f] = mean;
        vv[f] = rq[0] / cnt - mean * mean;
    }
}

__global__ void dbn_final(const float* mv, const float* vv, const float* g,
                          const float* bt, float* scale, float* shift) {
    int f = threadIdx.x;
    if (f >= V_N * 3) return;
    float s = g[f] * rsqrtf(vv[f] + EPSF);
    scale[f] = s;
    shift[f] = bt[f] - s * mv[f];
}

// BN + movement fields; writes f16 rows [m][32] (cols 0..2 data, 3..31 zero).
__global__ void preprocess_k(const float* __restrict__ pos,
                             const float* __restrict__ scale,
                             const float* __restrict__ shift,
                             f16* __restrict__ xo, f16* __restrict__ to,
                             f16* __restrict__ so) {
    long idx = (long)blockIdx.x * blockDim.x + threadIdx.x;
    if (idx >= MROWS) return;
    int b = (int)(idx / (T_LEN * V_N));
    int r = (int)(idx % (T_LEN * V_N));
    int t = r / V_N, v = r % V_N;
    int n = b >> 1, m = b & 1;
    int t2 = (t < T_LEN - 1) ? t + 1 : t - 1;
    float sgn = (t < T_LEN - 1) ? 1.0f : -1.0f;

    H8 px, pt, ps;
#pragma unroll
    for (int i = 0; i < 8; ++i) { px.h[i] = (f16)0.f; pt.h[i] = (f16)0.f; ps.h[i] = (f16)0.f; }
#pragma unroll
    for (int c = 0; c < 3; ++c) {
        int f = v * 3 + c;
        long base = (long)n * 32400 + (long)c * 10800;
        float xa = scale[f] * pos[base + t  * 36 + v * 2 + m] + shift[f];
        float xb = scale[f] * pos[base + t2 * 36 + v * 2 + m] + shift[f];
        int fc = 3 + c; // CENTER joint = 1
        float xc = scale[fc] * pos[base + t * 36 + 1 * 2 + m] + shift[fc];
        px.h[c] = (f16)xa;
        pt.h[c] = (f16)(sgn * (xb - xa));
        ps.h[c] = (f16)(xa - xc);
    }
    U4 z; z.x = z.y = z.z = z.w = 0u;
    U4* dx = (U4*)(xo + idx * 32);
    U4* dt = (U4*)(to + idx * 32);
    U4* ds = (U4*)(so + idx * 32);
    dx[0] = px.u; dx[1] = z; dx[2] = z; dx[3] = z;
    dt[0] = pt.u; dt[1] = z; dt[2] = z; dt[3] = z;
    ds[0] = ps.u; ds[1] = z; ds[2] = z; ds[3] = z;
}

// mask = (sum_k A > 0), a0n = row-normalized sum_k A   (18x18)
__global__ void atten_init(const float* __restrict__ A, float* __restrict__ a0n,
                           float* __restrict__ mask) {
    __shared__ float s[V_N * V_N];
    int i = threadIdx.x;
    if (i < V_N * V_N) {
        float v = A[i] + A[V_N * V_N + i] + A[2 * V_N * V_N + i];
        s[i] = v;
        mask[i] = (v > 0.f) ? 1.f : 0.f;
    }
    __syncthreads();
    if (i < V_N * V_N) {
        int vr = i / V_N;
        float sum = 0.f;
#pragma unroll
        for (int u = 0; u < V_N; ++u) sum += s[vr * V_N + u];
        a0n[i] = s[i] / sum;
    }
}

__global__ void atten_bcast(const float* __restrict__ a0n, float* __restrict__ atten,
                            long total) {
    long i = (long)blockIdx.x * blockDim.x + threadIdx.x;
    if (i < total) atten[i] = a0n[i % (V_N * V_N)];
}

// Per-channel stats over all rows (for residual-branch BN): LDS + global atomics.
__global__ void colstats_partial(const float* __restrict__ R, long mrows, int C,
                                 float* __restrict__ gsum, float* __restrict__ gsq) {
    __shared__ float ss[128], sq[128];
    if (threadIdx.x < C) { ss[threadIdx.x] = 0.f; sq[threadIdx.x] = 0.f; }
    __syncthreads();
    long total = mrows * C;
    for (long i = (long)blockIdx.x * 256 + threadIdx.x; i < total;
         i += (long)gridDim.x * 256) {
        float v = R[i];
        int c = (int)(i & (C - 1));        // C is a power of two (64/128)
        atomicAdd(&ss[c], v);
        atomicAdd(&sq[c], v * v);
    }
    __syncthreads();
    if (threadIdx.x < C) {
        atomicAdd(&gsum[threadIdx.x], ss[threadIdx.x]);
        atomicAdd(&gsq[threadIdx.x],  sq[threadIdx.x]);
    }
}

__global__ void colstats_final(const float* gsum, const float* gsq, long mrows,
                               const float* g, const float* bt,
                               float* asc, float* ash, int C) {
    int c = threadIdx.x;
    if (c >= C) return;
    float mean = gsum[c] / (float)mrows;
    float var  = gsq[c] / (float)mrows - mean * mean;
    float a = g[c] * rsqrtf(var + EPSF);
    asc[c] = a;
    ash[c] = bt[c] - a * mean;
}

// Fused: Ycomb = Yconv (+ a*R + c | + Xin), then out[v] = relu(sum_u att[v,u]*Ycomb[u]).
// One block per (b,t); res_mode: 1 = identity residual, 2 = conv+BN residual.
__global__ __launch_bounds__(256) void gcn_aggregate(
    const float* __restrict__ Y, const float* __restrict__ R,
    const float* __restrict__ asc, const float* __restrict__ ash,
    const f16* __restrict__ Xin, int ldx,
    const float* __restrict__ atten, int C,
    f16* __restrict__ out16, float* __restrict__ out32, int res_mode)
{
    __shared__ float sY[V_N * 128];
    __shared__ float sA[V_N * V_N];
    int bt = blockIdx.x;
    int tot = V_N * C;
    long rb = (long)bt * V_N;
    for (int i = threadIdx.x; i < tot; i += 256) {
        int u = i / C, c = i - u * C;
        long row = rb + u;
        float val = Y[row * C + c];
        if (res_mode == 2)      val += asc[c] * R[row * C + c] + ash[c];
        else if (res_mode == 1) val += (float)Xin[row * ldx + c];
        sY[i] = val;
    }
    for (int i = threadIdx.x; i < V_N * V_N; i += 256)
        sA[i] = atten[(long)bt * V_N * V_N + i];
    __syncthreads();
    for (int i = threadIdx.x; i < tot; i += 256) {
        int v = i / C, c = i - v * C;
        float acc = 0.f;
#pragma unroll
        for (int u = 0; u < V_N; ++u) acc += sA[v * V_N + u] * sY[u * C + c];
        float rr = fmaxf(acc, 0.f);
        long o = (rb + v) * C + c;
        out16[o] = (f16)rr;
        if (out32) out32[o] = rr;
    }
}

// s1/s2 = per-row dot of feat (64ch) with w1/w2.
__global__ void s12_kernel(const float* __restrict__ feat, const float* __restrict__ w1,
                           const float* __restrict__ w2, float* __restrict__ s1,
                           float* __restrict__ s2) {
    long m = (long)blockIdx.x * blockDim.x + threadIdx.x;
    if (m >= MROWS) return;
    const float* f = feat + m * 64;
    float a = 0.f, b = 0.f;
#pragma unroll
    for (int c = 0; c < 64; ++c) { float x = f[c]; a += w1[c] * x; b += w2[c] * x; }
    s1[m] = a; s2[m] = b;
}

// score = s1[v]+s2[u]+ba ; LeakyReLU(0.1) ; masked exp ; row normalize.
__global__ void atten_update(const float* __restrict__ s1, const float* __restrict__ s2,
                             const float* __restrict__ mask, const float* __restrict__ ba,
                             float* __restrict__ atten) {
    long idx = (long)blockIdx.x * blockDim.x + threadIdx.x;
    if (idx >= MROWS) return;
    long bt = idx / V_N;
    int v = (int)(idx % V_N);
    float b = ba[0];
    float sv = s1[idx];
    const float* s2r = s2 + bt * V_N;
    float e[V_N];
    float sum = 0.f;
#pragma unroll
    for (int u = 0; u < V_N; ++u) {
        float sc = sv + s2r[u] + b;
        sc = (sc > 0.f) ? sc : 0.1f * sc;
        float ev = expf(sc) * mask[v * V_N + u];
        e[u] = ev; sum += ev;
    }
    float inv = 1.0f / sum;
    float* o = atten + bt * (V_N * V_N) + (long)v * V_N;
#pragma unroll
    for (int u = 0; u < V_N; ++u) o[u] = e[u] * inv;
}

// Pool over (m,t,v) -> feat[n,128]. Rows for sample n are contiguous [n*10800, ...).
__global__ void pool_kernel(const float* __restrict__ P, float* __restrict__ feat) {
    __shared__ float sh[256];
    int n = blockIdx.x;
    int t = threadIdx.x;
    int c = t & 127, g = t >> 7;
    const float* base = P + (long)n * 10800 * 128;
    float s = 0.f;
    for (int r = g; r < 10800; r += 2) s += base[(long)r * 128 + c];
    sh[t] = s;
    __syncthreads();
    if (t < 128) feat[n * 128 + t] = (sh[t] + sh[t + 128]) * (1.0f / 10800.0f);
}

__global__ void fc_kernel(const float* __restrict__ feat, const float* __restrict__ w,
                          const float* __restrict__ b, float* __restrict__ out) {
    int i = blockIdx.x * blockDim.x + threadIdx.x;
    if (i >= N_BATCH * 400) return;
    int n = i / 400, o = i % 400;
    const float* f = feat + n * 128;
    const float* wr = w + (long)o * 128;
    float acc = b[o];
#pragma unroll 4
    for (int c = 0; c < 128; ++c) acc += f[c] * wr[c];
    out[i] = acc;
}

// ------------------------------------------------------------------ host ----
extern "C" void kernel_launch(void* const* d_in, const int* in_sizes, int n_in,
                              void* d_out, int out_size, void* d_ws, size_t ws_size,
                              hipStream_t stream) {
    (void)in_sizes; (void)n_in; (void)out_size; (void)ws_size;
    auto F = [&](int i) { return (const float*)d_in[i]; };

    const float* pos    = F(0);
    const float* Aadj   = F(1);
    const float* dbn_g  = F(2);
    const float* dbn_bt = F(3);

    const int cins[3]  = {3, 64, 64};
    const int couts[3] = {64, 64, 128};
    const int kpadIn[3] = {32, 64, 64};

    struct BlockP {
        const float *w[3], *b[3];
        const float *rw[3], *rb[3], *rg[3], *rbt[3];
        const float *wf, *bf, *w1, *w2, *ba;
        bool has_res;
    } blk[3];

    int idx = 4;
    for (int b = 0; b < 3; ++b) {
        bool hr = (cins[b] != couts[b]);
        blk[b].has_res = hr;
        for (int s = 0; s < 3; ++s) {           // dict order: pos, tmf, smf
            blk[b].w[s] = F(idx++); blk[b].b[s] = F(idx++);
            if (hr) {
                blk[b].rw[s]  = F(idx++); blk[b].rb[s]  = F(idx++);
                blk[b].rg[s]  = F(idx++); blk[b].rbt[s] = F(idx++);
            } else {
                blk[b].rw[s] = blk[b].rb[s] = blk[b].rg[s] = blk[b].rbt[s] = nullptr;
            }
        }
        blk[b].wf = F(idx++); blk[b].bf = F(idx++);
        blk[b].w1 = F(idx++); blk[b].w2 = F(idx++); blk[b].ba = F(idx++);
    }
    const float* fcw = F(idx++);
    const float* fcb = F(idx++);

    // ---- workspace carve-up (deterministic) ----
    char* base = (char*)d_ws;
    size_t cur = 0;
    auto alloc = [&](size_t bytes) -> void* {
        void* p = base + cur;
        cur = (cur + bytes + 255) & ~(size_t)255;
        return p;
    };
    f16* sA[3]; f16* sB[3];
    for (int s = 0; s < 3; ++s) sA[s] = (f16*)alloc((size_t)MROWS * 64 * 2);
    for (int s = 0; s < 3; ++s) sB[s] = (f16*)alloc((size_t)MROWS * 128 * 2);
    float* Ybuf   = (float*)alloc((size_t)MROWS * 128 * 4);
    float* Rbuf   = (float*)alloc((size_t)MROWS * 128 * 4);   // also aliased as feat
    float* posF32 = (float*)alloc((size_t)MROWS * 128 * 4);
    float* atten  = (float*)alloc((size_t)BT_TOT * V_N * V_N * 4);
    float* s1buf  = (float*)alloc((size_t)MROWS * 4);
    float* s2buf  = (float*)alloc((size_t)MROWS * 4);
    float* statsum = (float*)alloc(256 * 4);
    float* ascbuf  = (float*)alloc(128 * 4);
    float* ashbuf  = (float*)alloc(128 * 4);
    float* dbn_mean  = (float*)alloc(64 * 4);
    float* dbn_var   = (float*)alloc(64 * 4);
    float* dbn_scale = (float*)alloc(64 * 4);
    float* dbn_shift = (float*)alloc(64 * 4);
    float* a0n     = (float*)alloc(V_N * V_N * 4);
    float* maskbuf = (float*)alloc(V_N * V_N * 4);
    float* featN   = (float*)alloc(N_BATCH * 128 * 4);
    float* featbuf = Rbuf;  // feat (M x 64) reuses R scratch between gcn passes

    unsigned int *pk_main[3][3], *pk_res[3][3], *pk_wf1[3], *pk_wf2[3];
    for (int b = 0; b < 3; ++b) {
        for (int s = 0; s < 3; ++s) {
            pk_main[b][s] = (unsigned int*)alloc((size_t)kpadIn[b] * couts[b] * 2);
            pk_res[b][s]  = blk[b].has_res
                ? (unsigned int*)alloc((size_t)kpadIn[b] * couts[b] * 2) : nullptr;
        }
        pk_wf1[b] = (unsigned int*)alloc((size_t)couts[b] * 64 * 2);
        pk_wf2[b] = (unsigned int*)alloc((size_t)couts[b] * 64 * 2);
    }

    // ---- pack weights into WMMA B-fragment order ----
    auto packW = [&](const float* W, int ldw, int col0, int cvalid, int kpad, int n,
                     unsigned int* dst) {
        int total = (kpad / 32) * (n / 16) * 32;
        pack_wmma_b<<<(total + 255) / 256, 256, 0, stream>>>(W, ldw, col0, cvalid,
                                                             kpad, n, dst);
    };
    for (int b = 0; b < 3; ++b) {
        for (int s = 0; s < 3; ++s) {
            packW(blk[b].w[s], cins[b], 0, cins[b], kpadIn[b], couts[b], pk_main[b][s]);
            if (blk[b].has_res)
                packW(blk[b].rw[s], cins[b], 0, cins[b], kpadIn[b], couts[b], pk_res[b][s]);
        }
        packW(blk[b].wf, 2 * couts[b], 0,        couts[b], couts[b], 64, pk_wf1[b]);
        packW(blk[b].wf, 2 * couts[b], couts[b], couts[b], couts[b], 64, pk_wf2[b]);
    }

    // ---- data_bn + preprocessing + initial attention ----
    dbn_stats<<<V_N * 3, 256, 0, stream>>>(pos, dbn_mean, dbn_var);
    dbn_final<<<1, 64, 0, stream>>>(dbn_mean, dbn_var, dbn_g, dbn_bt, dbn_scale, dbn_shift);
    preprocess_k<<<(int)((MROWS + 255) / 256), 256, 0, stream>>>(
        pos, dbn_scale, dbn_shift, sA[0], sA[1], sA[2]);
    atten_init<<<1, 352, 0, stream>>>(Aadj, a0n, maskbuf);
    {
        long tot = (long)BT_TOT * V_N * V_N;
        atten_bcast<<<(int)((tot + 255) / 256), 256, 0, stream>>>(a0n, atten, tot);
    }

    auto gemm = [&](const f16* Ap, int lda, const unsigned int* Bp, const float* bias,
                    float* Cp, int ldc, int kpad, int n, int accum) {
        dim3 g((unsigned)(MROWS / 256), (unsigned)(n / 64));
        wmma_gemm_f16<<<g, 256, 0, stream>>>(Ap, lda, (const U4*)Bp, bias, Cp, ldc,
                                             kpad, n, accum);
    };

    f16** cur_in = sA;
    f16** cur_out = sB;

    for (int b = 0; b < 3; ++b) {
        int cout = couts[b];
        int lda  = kpadIn[b];

        auto run_gcn = [&](int s, bool wf32) {
            gemm(cur_in[s], lda, pk_main[b][s], blk[b].b[s], Ybuf, cout, lda, cout, 0);
            int res_mode;
            if (blk[b].has_res) {
                gemm(cur_in[s], lda, pk_res[b][s], blk[b].rb[s], Rbuf, cout, lda, cout, 0);
                zero_f32<<<1, 256, 0, stream>>>(statsum, 2 * cout);
                colstats_partial<<<512, 256, 0, stream>>>(Rbuf, MROWS, cout,
                                                          statsum, statsum + cout);
                colstats_final<<<1, 128, 0, stream>>>(statsum, statsum + cout, MROWS,
                                                      blk[b].rg[s], blk[b].rbt[s],
                                                      ascbuf, ashbuf, cout);
                res_mode = 2;
            } else {
                res_mode = 1;
            }
            gcn_aggregate<<<BT_TOT, 256, 0, stream>>>(
                Ybuf, Rbuf, ascbuf, ashbuf, cur_in[s], lda, atten, cout,
                cur_out[s], wf32 ? posF32 : nullptr, res_mode);
        };

        run_gcn(1, false);   // tmf (old attention)
        run_gcn(2, false);   // smf (old attention)

        // attention generator (uses NEW tmf/smf features)
        gemm(cur_out[1], cout, pk_wf1[b], blk[b].bf, featbuf, 64, cout, 64, 0);
        gemm(cur_out[2], cout, pk_wf2[b], nullptr,   featbuf, 64, cout, 64, 1);
        s12_kernel<<<(int)((MROWS + 255) / 256), 256, 0, stream>>>(featbuf, blk[b].w1,
                                                                   blk[b].w2, s1buf, s2buf);
        atten_update<<<(int)((MROWS + 255) / 256), 256, 0, stream>>>(s1buf, s2buf,
                                                                     maskbuf, blk[b].ba,
                                                                     atten);

        run_gcn(0, b == 2);  // pos (new attention); last block also writes f32

        f16** tmp = cur_in; cur_in = cur_out; cur_out = tmp;
    }

    pool_kernel<<<N_BATCH, 256, 0, stream>>>(posF32, featN);
    fc_kernel<<<(N_BATCH * 400 + 255) / 256, 256, 0, stream>>>(featN, fcw, fcb,
                                                               (float*)d_out);
}
